// AttentionLayer_32650341384848
// MI455X (gfx1250) — matrix operand order repaired and checked by reading, not compiled, not run on hardware
//
#include <hip/hip_runtime.h>
#include <hip/hip_bf16.h>

// ---------------------------------------------------------------------------
// Fused causal attention for MI455X (gfx1250, wave32, WMMA 16x16x32 f16).
//   b=4, s=4096, d_model=1024, d_k=d_v=64, fp32 in/out, f16 matrix operands
//   with fp32 accumulation.
// Pipeline: pack_weights -> qkv_proj (TDM double-buffered LDS WMMA GEMM)
//           -> attn (flash, WMMA, register-pipelined K/V).
// ---------------------------------------------------------------------------

typedef __attribute__((ext_vector_type(16))) _Float16 v16h;
typedef __attribute__((ext_vector_type(8)))  float    v8f;
typedef __attribute__((ext_vector_type(4)))  unsigned u32x4;
typedef __attribute__((ext_vector_type(8)))  unsigned u32x8;

#define D_MODEL 1024
#define D_K     64
#define D_V     64
#define SEQ     4096
#define BATCH   4
#define R_TOT   (BATCH * SEQ)          // 16384 total rows

// Workspace layout (halves):
//   wB : 32 chunks * 12 frags * 512 = 196608  (chunk-major f16 B-fragments)
//   qh : R_TOT * 64                  (Q, row-major f16)
//   kh : R_TOT * 64                  (K, row-major f16)
//   vt : 64 * R_TOT                  (V transposed: [dv][row], f16)
#define WB_HALVES  (32 * 12 * 512)
#define QKV_HALVES ((size_t)R_TOT * 64)
#define CHUNK_BYTES (12 * 512 * 2)     // 12288 B of B-fragments per K-chunk

static __device__ inline v16h load_v16h(const _Float16* p) {
  union { v16h h; uint4 u[2]; } u;
  u.u[0] = *(const uint4*)(p);
  u.u[1] = *(const uint4*)(p + 8);
  return u.h;
}

// lane-pair exchange (lanes l <-> l^16): ds_swizzle SWAPX16 (xor=0x10,and=0x1f)
static __device__ inline float swap16(float x) {
  return __int_as_float(__builtin_amdgcn_ds_swizzle(__float_as_int(x), 0x401F));
}

static __device__ inline v8f wmma_f16(v16h a, v16h b, v8f c) {
  return __builtin_amdgcn_wmma_f32_16x16x32_f16(
      /*neg_a=*/false, a, /*neg_b=*/false, b,
      /*c_mod=*/(short)0, c, /*reuse_a=*/false, /*reuse_b=*/false);
}

// ---------------------------------------------------------------------------
// TDM: issue one tensor_load_to_lds moving `bytes` contiguous bytes
// (bytes % 8 == 0) from global `gaddr` to LDS offset `lds_addr`.
// D# per CDNA5 ISA ch.8: group0 = {count, lds_addr, global_addr, type=2},
// group1 = {data_size=8B, tensor_dim0 = tile_dim0 = bytes/8, dim1 = 1}.
// Tracked with TENSORcnt (in-order per wave).
// ---------------------------------------------------------------------------
static __device__ inline void tdm_load_chunk(const void* gaddr,
                                             unsigned lds_addr,
                                             unsigned elems8 /* bytes/8 */) {
  const unsigned long long ga = (unsigned long long)gaddr;
  u32x4 g0;
  g0[0] = 1u;                                    // count=1, is_restore=0
  g0[1] = lds_addr;                              // lds_addr [63:32]
  g0[2] = (unsigned)ga;                          // global_addr low
  g0[3] = ((unsigned)(ga >> 32) & 0x01FFFFFFu)   // global_addr[56:32]
          | (2u << 30);                          // type=2 ("image")
  u32x8 g1;
  g1[0] = 3u << 16;                              // data_size=3 (8 bytes)
  g1[1] = (elems8 & 0xFFFFu) << 16;              // tensor_dim0[15:0]
  g1[2] = ((elems8 >> 16) & 0xFFFFu)             // tensor_dim0[31:16]
          | (1u << 16);                          // tensor_dim1 = 1
  g1[3] = (elems8 & 0xFFFFu) << 16;              // tile_dim0 (tensor_dim1 hi=0)
  g1[4] = 1u;                                    // tile_dim1 = 1, tile_dim2 = 0
  g1[5] = elems8;                                // tensor_dim0_stride low
  g1[6] = 0u;                                    // stride hi / dim1_stride lo
  g1[7] = 0u;                                    // dim1_stride hi
  asm volatile("tensor_load_to_lds %0, %1" :: "s"(g0), "s"(g1) : "memory");
}

// ---------------------------------------------------------------------------
// Kernel 1: pre-swizzle fp32 weights into WMMA B-fragment layout (f16),
// CHUNK-MAJOR: fragment index f = ((kc*3 + mtx)*4 + nt) so one K-chunk's
// 12 fragments (12 KB) are contiguous for the TDM staging.
// B layout (16-bit, 32x16): lanes 0-15 hold K=0..7,16..23 of N=lane;
//                           lanes 16-31 hold K=8..15,24..31 of N=lane-16.
// ---------------------------------------------------------------------------
__global__ void pack_weights(const float* __restrict__ wq,
                             const float* __restrict__ wk,
                             const float* __restrict__ wv,
                             _Float16* __restrict__ wB) {
  const int f = blockIdx.x;          // 0..383, chunk-major
  const int l = threadIdx.x;         // 0..31
  const int kc  = f / 12;
  const int r   = f % 12;
  const int mtx = r >> 2;
  const int nt  = r & 3;
  const float* w = (mtx == 0) ? wq : ((mtx == 1) ? wk : wv);
  const int kb = kc * 32 + ((l & 16) ? 8 : 0);
  const int n  = nt * 16 + (l & 15);
  _Float16* dst = wB + (size_t)f * 512 + l * 16;
#pragma unroll
  for (int p = 0; p < 16; ++p)
    dst[p] = (_Float16)w[(size_t)(kb + p + (p >= 8 ? 8 : 0)) * D_K + n];
}

// ---------------------------------------------------------------------------
// Kernel 2: QKV projection. 8 waves/block, one 16-row tile each.
// Weight fragments double-buffered in LDS; each 12 KB chunk fetched by ONE
// tensor_load_to_lds (wave 0), overlapped with compute of the previous chunk.
// Fragment ds_loads are register-pipelined against the 12 WMMAs.
// A frag (f16 16x32): lanes 0-15: K in {0..7,16..23}; lanes 16-31: {8..15,24..31}.
// ---------------------------------------------------------------------------
__global__ __launch_bounds__(256) void qkv_proj(
    const float* __restrict__ x, const _Float16* __restrict__ wB,
    const float* __restrict__ bq, const float* __restrict__ bk,
    const float* __restrict__ bv,
    _Float16* __restrict__ qh, _Float16* __restrict__ kh,
    _Float16* __restrict__ vt) {
  __shared__ __align__(16) _Float16 sB[2][12 * 512];   // 2 x 12 KB

  const int lane = threadIdx.x & 31;
  const int wvb  = threadIdx.x >> 5;                   // wave in block
  const int wave = blockIdx.x * (blockDim.x >> 5) + wvb;
  const int m0 = wave * 16;
  const bool hi = lane >= 16;
  const int ln = lane & 15;

  const unsigned ldsb[2] = {
      (unsigned)(unsigned long long)(&sB[0][0]),
      (unsigned)(unsigned long long)(&sB[1][0])};
  const char* wBb = (const char*)wB;

  v8f acc[12];                      // f = mtx*4 + nt
#pragma unroll
  for (int f = 0; f < 12; ++f)
#pragma unroll
    for (int i = 0; i < 8; ++i) acc[f][i] = 0.0f;

  const float* xrow = x + (size_t)(m0 + ln) * D_MODEL;

  // TDM prefetch of chunk 0 into buffer 0 (one DMA for the whole block).
  if (wvb == 0)
    tdm_load_chunk(wBb, ldsb[0], CHUNK_BYTES / 8);

  for (int kc = 0; kc < 32; ++kc) {
    if (wvb == 0) {
      if (kc + 1 < 32) {
        tdm_load_chunk(wBb + (size_t)(kc + 1) * CHUNK_BYTES,
                       ldsb[(kc + 1) & 1], CHUNK_BYTES / 8);
        __builtin_amdgcn_s_wait_tensorcnt(1);   // chunk kc complete (in-order)
      } else {
        __builtin_amdgcn_s_wait_tensorcnt(0);
      }
    }
    __syncthreads();   // current chunk visible to all waves

    // A fragment from x (fp32 -> f16 in-register).
    const int kb = kc * 32 + (hi ? 8 : 0);
    float4 f0 = *(const float4*)(xrow + kb);
    float4 f1 = *(const float4*)(xrow + kb + 4);
    float4 f2 = *(const float4*)(xrow + kb + 16);
    float4 f3 = *(const float4*)(xrow + kb + 20);
    v16h a;
    a[0]=(_Float16)f0.x; a[1]=(_Float16)f0.y; a[2]=(_Float16)f0.z; a[3]=(_Float16)f0.w;
    a[4]=(_Float16)f1.x; a[5]=(_Float16)f1.y; a[6]=(_Float16)f1.z; a[7]=(_Float16)f1.w;
    a[8]=(_Float16)f2.x; a[9]=(_Float16)f2.y; a[10]=(_Float16)f2.z; a[11]=(_Float16)f2.w;
    a[12]=(_Float16)f3.x; a[13]=(_Float16)f3.y; a[14]=(_Float16)f3.z; a[15]=(_Float16)f3.w;

    // 12 WMMAs with register-double-buffered fragment loads:
    // ds_load of fragment f+1 overlaps the WMMA of fragment f.
    const _Float16* sbuf = sB[kc & 1];
    v16h bcur = load_v16h(sbuf + lane * 16);
#pragma unroll
    for (int f = 0; f < 12; ++f) {
      v16h bnxt;
      if (f + 1 < 12) bnxt = load_v16h(sbuf + (f + 1) * 512 + lane * 16);
      acc[f] = wmma_f16(a, bcur, acc[f]);
      bcur = bnxt;
    }
    __syncthreads();   // done reading this buffer before TDM overwrites it
  }

  // bias + store. C layout: element (VGPR i, lane): row = m0+i+(hi?8:0),
  // col n = nt*16 + (lane&15).
  const float* bias[3] = {bq, bk, bv};
#pragma unroll
  for (int mtx = 0; mtx < 3; ++mtx) {
#pragma unroll
    for (int nt = 0; nt < 4; ++nt) {
      const float bb = bias[mtx][nt * 16 + ln];
      v8f c = acc[mtx * 4 + nt];
#pragma unroll
      for (int i = 0; i < 8; ++i) c[i] += bb;
      if (mtx < 2) {
        _Float16* dst = (mtx == 0) ? qh : kh;
#pragma unroll
        for (int i = 0; i < 8; ++i)
          dst[(size_t)(m0 + i + (hi ? 8 : 0)) * 64 + nt * 16 + ln] =
              (_Float16)c[i];
      } else {
        // vt[dv][row]: contiguous along VGPR index -> pack half-pairs.
        const size_t base = (size_t)(nt * 16 + ln) * R_TOT + m0 + (hi ? 8 : 0);
#pragma unroll
        for (int i = 0; i < 8; i += 2) {
          union { _Float16 h[2]; unsigned u; } pk;
          pk.h[0] = (_Float16)c[i];
          pk.h[1] = (_Float16)c[i + 1];
          *(unsigned*)(vt + base + i) = pk.u;
        }
      }
    }
  }
}

// ---------------------------------------------------------------------------
// Kernel 3: flash attention (causal). One wave per (batch, 16-query tile).
// S^T = K*Q^T (softmax axis in-lane + one SWAPX16), O^T = V^T*P^T.
// K fragments register-double-buffered across key blocks; V fragments issued
// at the top of the iteration and consumed after softmax (latency hidden).
// ---------------------------------------------------------------------------
static __device__ inline void load_kfrags(const _Float16* __restrict__ kh,
                                          size_t gk, int ln, int off,
                                          v16h* out) {
#pragma unroll
  for (int st = 0; st < 2; ++st) {
    const _Float16* kp = kh + (gk + st * 16 + ln) * 64;
    union { v16h h; uint4 u[2]; } u;
    u.u[0] = *(const uint4*)(kp + 2 * off);
    u.u[1] = *(const uint4*)(kp + 8 + 2 * off);
    out[st * 2 + 0] = u.h;
    u.u[0] = *(const uint4*)(kp + 32 + 2 * off);
    u.u[1] = *(const uint4*)(kp + 40 + 2 * off);
    out[st * 2 + 1] = u.h;
  }
}

__global__ __launch_bounds__(256) void attn(
    const _Float16* __restrict__ qh, const _Float16* __restrict__ kh,
    const _Float16* __restrict__ vt, float* __restrict__ out) {
  const int lane = threadIdx.x & 31;
  const int wave = blockIdx.x * (blockDim.x >> 5) + (threadIdx.x >> 5);
  const int b  = wave >> 8;          // batch
  const int qt = wave & 255;         // 16-query tile within batch
  const bool hi = lane >= 16;
  const int ln = lane & 15;
  const int off = hi ? 8 : 0;

  const int    qlocal = qt * 16 + ln;                 // query index in batch
  const size_t qrow   = (size_t)b * SEQ + qlocal;     // global row

  // Q as B-fragments (loop invariant).
  const v16h qb0 = load_v16h(qh + qrow * 64 + (hi ? 16 : 0));
  const v16h qb1 = load_v16h(qh + qrow * 64 + 32 + (hi ? 16 : 0));

  v8f o[4];
#pragma unroll
  for (int t = 0; t < 4; ++t)
#pragma unroll
    for (int i = 0; i < 8; ++i) o[t][i] = 0.0f;

  float m_run = -1e30f, l_run = 0.0f;
  const int nblk = (qt >> 1) + 1;    // 32-key blocks covering keys <= q_hi

  v16h kf[4];                        // current block's K fragments
  load_kfrags(kh, (size_t)b * SEQ, ln, off, kf);

  for (int jt = 0; jt < nblk; ++jt) {
    const int    kb = jt * 32;
    const size_t gk = (size_t)b * SEQ + kb;

    // V fragments for this block: issue now, consume after softmax.
    v16h vf[4];
#pragma unroll
    for (int t = 0; t < 4; ++t) {
      const _Float16* vp = vt + (size_t)(t * 16 + ln) * R_TOT + gk;
      union { v16h h; uint4 u[2]; } u;
      u.u[0] = *(const uint4*)(vp + 2 * off);
      u.u[1] = *(const uint4*)(vp + 8 + 2 * off);
      vf[t] = u.h;
    }

    // Prefetch next block's K fragments.
    v16h kn[4];
    if (jt + 1 < nblk)
      load_kfrags(kh, gk + 32, ln, off, kn);

    // ---- S^T tiles: two 16-key tiles, dk=64 in two WMMA chunks ------------
    v8f s[2];
#pragma unroll
    for (int st = 0; st < 2; ++st) {
      v8f c;
#pragma unroll
      for (int i = 0; i < 8; ++i) c[i] = 0.0f;
      c = wmma_f16(kf[st * 2 + 0], qb0, c);
      c = wmma_f16(kf[st * 2 + 1], qb1, c);
      s[st] = c;
    }

    // ---- scale, causal mask, online softmax -------------------------------
    float tmax = -1e30f;
#pragma unroll
    for (int st = 0; st < 2; ++st)
#pragma unroll
      for (int i = 0; i < 8; ++i) {
        const int key = kb + st * 16 + i + off;
        float v = s[st][i] * 0.125f;               // 1/sqrt(64)
        if (key > qlocal) v = -1e30f;
        s[st][i] = v;
        tmax = fmaxf(tmax, v);
      }
    tmax = fmaxf(tmax, swap16(tmax));
    const float m_new = fmaxf(m_run, tmax);
    const float scale_old = __expf(m_run - m_new);
    float lsum = 0.0f;
#pragma unroll
    for (int st = 0; st < 2; ++st)
#pragma unroll
      for (int i = 0; i < 8; ++i) {
        const float p = __expf(s[st][i] - m_new);
        s[st][i] = p;
        lsum += p;
      }
    lsum += swap16(lsum);
    l_run = l_run * scale_old + lsum;
    m_run = m_new;
#pragma unroll
    for (int t = 0; t < 4; ++t)
#pragma unroll
      for (int i = 0; i < 8; ++i) o[t][i] *= scale_old;

    // ---- P^T as B-fragment: one swizzle per VGPR to merge lane halves -----
    v16h pb;
#pragma unroll
    for (int i = 0; i < 8; ++i) {
      const float oth0 = swap16(s[0][i]);
      const float oth1 = swap16(s[1][i]);
      const float e_lo = hi ? oth1 : s[0][i];
      const float e_hi = hi ? s[1][i] : oth0;
      pb[i]     = (_Float16)e_lo;
      pb[8 + i] = (_Float16)e_hi;
    }

    // ---- O^T += V^T * P^T (4 dv tiles) ------------------------------------
#pragma unroll
    for (int t = 0; t < 4; ++t)
      o[t] = wmma_f16(vf[t], pb, o[t]);

    // rotate K double-buffer
#pragma unroll
    for (int t = 0; t < 4; ++t) kf[t] = kn[t];
  }

  // ---- normalize and store: dv contiguous along VGPR index -> float4 -----
  const float inv = 1.0f / l_run;
  float* op = out + qrow * 64 + off;
#pragma unroll
  for (int t = 0; t < 4; ++t) {
    float4 lo, h4;
    lo.x = o[t][0] * inv; lo.y = o[t][1] * inv;
    lo.z = o[t][2] * inv; lo.w = o[t][3] * inv;
    h4.x = o[t][4] * inv; h4.y = o[t][5] * inv;
    h4.z = o[t][6] * inv; h4.w = o[t][7] * inv;
    *(float4*)(op + t * 16)     = lo;
    *(float4*)(op + t * 16 + 4) = h4;
  }
}

// ---------------------------------------------------------------------------
extern "C" void kernel_launch(void* const* d_in, const int* in_sizes, int n_in,
                              void* d_out, int out_size, void* d_ws,
                              size_t ws_size, hipStream_t stream) {
  const float* x  = (const float*)d_in[0];
  const float* wq = (const float*)d_in[1];
  const float* bq = (const float*)d_in[2];
  const float* wk = (const float*)d_in[3];
  const float* bk = (const float*)d_in[4];
  const float* wv = (const float*)d_in[5];
  const float* bv = (const float*)d_in[6];
  float* out = (float*)d_out;

  _Float16* wB = (_Float16*)d_ws;                 // ~384 KB
  _Float16* qh = wB + WB_HALVES;                  // 2 MB
  _Float16* kh = qh + QKV_HALVES;                 // 2 MB
  _Float16* vt = kh + QKV_HALVES;                 // 2 MB  (total ~6.4 MB)

  pack_weights<<<32 * 12, 32, 0, stream>>>(wq, wk, wv, wB);
  qkv_proj<<<R_TOT / 16 / 8, 256, 0, stream>>>(x, wB, bq, bk, bv, qh, kh, vt);
  attn<<<BATCH * (SEQ / 16) / 8, 256, 0, stream>>>(qh, kh, vt, out);
}
